// SpatialAttnProcessor2_0_85487029060028
// MI455X (gfx1250) — matrix operand order repaired and checked by reading, not compiled, not run on hardware
//
#include <hip/hip_runtime.h>

// ---------------- problem constants ----------------
#define CHANNELS   1280
#define NUM_TOKENS 576
#define IMG_COUNT  5
#define HEADS      20
#define HD         64
#define N_SEL      288
#define MTOT       (2*IMG_COUNT*NUM_TOKENS)   // 5760 total tokens
#define TK         1728                       // enc length per (b,i)
#define OTHERS     (4*N_SEL)                  // 1152 gathered tokens

typedef __attribute__((ext_vector_type(16))) __bf16 v16bf;
typedef __attribute__((ext_vector_type(8)))  float  v8f;
typedef __attribute__((address_space(3))) char lds_char;

union BFrag { v16bf v; uint4 q[2]; unsigned short s[16]; };

__device__ __forceinline__ unsigned short f2bf(float f) {
  union { float f; unsigned u; } x; x.f = f;
  unsigned r = x.u + 0x7FFFu + ((x.u >> 16) & 1u);   // round-to-nearest-even
  return (unsigned short)(r >> 16);
}

__device__ __forceinline__ v8f wmma_bf16(const v16bf& a, const v16bf& b, v8f c) {
  return __builtin_amdgcn_wmma_f32_16x16x32_bf16(false, a, false, b, (short)0, c,
                                                 false, false);
}

// LDS byte offset of a generic pointer into a __shared__ object
__device__ __forceinline__ unsigned ldsAddrOf(const void* p) {
  return (unsigned)(unsigned long long)(const lds_char*)p;
}

// one 16-byte async copy global -> LDS (ASYNCcnt-tracked, bypasses VGPRs)
__device__ __forceinline__ void asyncLoadB128(unsigned lds, const void* g) {
  asm volatile("global_load_async_to_lds_b128 %0, %1, off"
               :: "v"(lds), "v"((unsigned long long)g) : "memory");
}

// ---------------------------------------------------------------------------
// elementwise f32 -> bf16 (one-time conversion of activations and weights)
// ---------------------------------------------------------------------------
__global__ void f32_to_bf16_kernel(const float* __restrict__ src,
                                   unsigned short* __restrict__ dst, int n) {
  int i = blockIdx.x * blockDim.x + threadIdx.x;
  const int stride = gridDim.x * blockDim.x;
  for (; i < n; i += stride) dst[i] = f2bf(src[i]);
}

// ---------------------------------------------------------------------------
// GEMM: C[M,N] = A[M,K](bf16) @ W[N,K]^T(bf16), f32 accum.
// Double-buffered async-to-LDS staging; 8 waves (4M x 2N), wave = 32x64.
// Output either bf16 (outb, scaled) or f32 (outf, + bias).
// ---------------------------------------------------------------------------
#define BM  128
#define BN  128
#define BK  32
#define LDT 40     // padded LDS k-stride (elements); rows 80B -> 16B-aligned runs

__global__ __launch_bounds__(256, 2)
void gemm_bf16_async(const unsigned short* __restrict__ A,
                     const unsigned short* __restrict__ W,
                     unsigned short* __restrict__ outb, float* __restrict__ outf,
                     const float* __restrict__ bias, float scaleOut,
                     int K, int N)
{
  __shared__ __align__(16) unsigned short As[2][BM * LDT];
  __shared__ __align__(16) unsigned short Bs[2][BN * LDT];

  const int t    = threadIdx.x;
  const int m0   = blockIdx.x * BM;
  const int n0   = blockIdx.y * BN;
  const int lane = t & 31, wid = t >> 5;
  const int lrow = lane & 15;
  const bool hiL = lane >= 16;
  const int wm   = (wid & 3) * 32;
  const int wn   = (wid >> 2) * 64;

  const unsigned aBase[2] = { ldsAddrOf(&As[0][0]), ldsAddrOf(&As[1][0]) };
  const unsigned bBase[2] = { ldsAddrOf(&Bs[0][0]), ldsAddrOf(&Bs[1][0]) };

  v8f acc[2][4];
#pragma unroll
  for (int mi = 0; mi < 2; ++mi)
#pragma unroll
    for (int ni = 0; ni < 4; ++ni)
#pragma unroll
      for (int v = 0; v < 8; ++v) acc[mi][ni][v] = 0.0f;

  // tile = 128 rows x 32 k of bf16 = 512 x 16B chunks; 256 threads -> 2 chunks
  // each for A and for B => 4 async ops per thread per tile-pair.
  auto issueTile = [&](int bb, int k0) {
    for (int c = t; c < 512; c += 256) {
      const int row = c >> 2, ch = c & 3;
      const unsigned loff = (unsigned)(row * (LDT * 2) + ch * 16);
      asyncLoadB128(aBase[bb] + loff, A + (size_t)(m0 + row) * K + k0 + ch * 8);
      asyncLoadB128(bBase[bb] + loff, W + (size_t)(n0 + row) * K + k0 + ch * 8);
    }
  };

  issueTile(0, 0);
  const int nk = K / BK;
  for (int ki = 0; ki < nk; ++ki) {
    const int cur = ki & 1;
    if (ki + 1 < nk) {
      issueTile(cur ^ 1, (ki + 1) * BK);
      // async loads complete in order: <=4 outstanding => current tile landed
      asm volatile("s_wait_asynccnt 4" ::: "memory");
    } else {
      asm volatile("s_wait_asynccnt 0" ::: "memory");
    }
    __syncthreads();

    BFrag af[2], bfr[4];
    const int ka = hiL ? 16 : 0;   // A lane k-run: 16 contiguous
#pragma unroll
    for (int mi = 0; mi < 2; ++mi) {
      const unsigned short* p = &As[cur][(wm + mi * 16 + lrow) * LDT];
      af[mi].q[0] = *(const uint4*)(p + ka);
      af[mi].q[1] = *(const uint4*)(p + ka + 8);
    }
    const int kb = hiL ? 16 : 0;   // B lane k-run: 16 contiguous
#pragma unroll
    for (int ni = 0; ni < 4; ++ni) {
      const unsigned short* p = &Bs[cur][(wn + ni * 16 + lrow) * LDT + kb];
      bfr[ni].q[0] = *(const uint4*)(p);
      bfr[ni].q[1] = *(const uint4*)(p + 8);
    }
#pragma unroll
    for (int mi = 0; mi < 2; ++mi)
#pragma unroll
      for (int ni = 0; ni < 4; ++ni)
        acc[mi][ni] = wmma_bf16(af[mi].v, bfr[ni].v, acc[mi][ni]);

    __syncthreads();   // all waves done with buf[cur] before it is re-filled
  }

  // epilogue: C layout = rows in VGPRs (v + hiL*8), cols in lanes (lrow)
#pragma unroll
  for (int mi = 0; mi < 2; ++mi) {
    const int rb = m0 + wm + mi * 16 + (hiL ? 8 : 0);
#pragma unroll
    for (int ni = 0; ni < 4; ++ni) {
      const int col = n0 + wn + ni * 16 + lrow;
#pragma unroll
      for (int v = 0; v < 8; ++v) {
        const float val = acc[mi][ni][v] * scaleOut;
        const size_t off = (size_t)(rb + v) * N + col;
        if (outb) outb[off] = f2bf(val);
        else      outf[off] = val + bias[col];
      }
    }
  }
}

// ---------------------------------------------------------------------------
// Flash attention over gathered K/V. Block = 64 q-rows x 1 head x (b,i).
// 4 waves; wave w owns q-rows [w*16, w*16+16). Tk in blocks of 32.
// K rows staged async-to-LDS; V transposed through VGPRs. Output bf16.
// ---------------------------------------------------------------------------
__global__ __launch_bounds__(128, 2)
void attn_kernel(const unsigned short* __restrict__ Qb,
                 const unsigned short* __restrict__ Kb,
                 const unsigned short* __restrict__ Vb,
                 unsigned short* __restrict__ Ob,
                 const long long* __restrict__ indices)
{
  __shared__ __align__(16) unsigned short Ks[32 * 72];      // K rows (j, d)
  __shared__ __align__(16) unsigned short Vt[64 * 40];      // V transposed (d, j)
  __shared__ __align__(16) unsigned short Pw[4 * 16 * 40];  // per-wave P repack

  const int t    = threadIdx.x;
  const int qblk = blockIdx.x;
  const int h    = blockIdx.y;
  const int bi   = blockIdx.z;
  const int b    = bi / IMG_COUNT;
  const int img  = bi % IMG_COUNT;
  const int lane = t & 31, wid = t >> 5;
  const int lrow = lane & 15;
  const bool hiL = lane >= 16;
  const int ka   = hiL ? 16 : 0;
  const unsigned ksBase = ldsAddrOf(&Ks[0]);

  // ---- Q strip (16 rows x 64 dims) as two A fragments (already *0.125) ----
  const size_t qrow = (size_t)bi * NUM_TOKENS + qblk * 64 + wid * 16 + lrow;
  const unsigned short* qsrc = Qb + qrow * CHANNELS + (size_t)h * HD;
  BFrag qa[2];
  qa[0].q[0] = *(const uint4*)(qsrc + ka);
  qa[0].q[1] = *(const uint4*)(qsrc + ka + 8);
  qa[1].q[0] = *(const uint4*)(qsrc + 32 + ka);
  qa[1].q[1] = *(const uint4*)(qsrc + 32 + ka + 8);

  v8f acc[4];
  float m[8], l[8];
#pragma unroll
  for (int v = 0; v < 8; ++v) { m[v] = -1e30f; l[v] = 0.0f; }
#pragma unroll
  for (int dt = 0; dt < 4; ++dt)
#pragma unroll
    for (int v = 0; v < 8; ++v) acc[dt][v] = 0.0f;

  const int jrow = t >> 2;   // 0..31 : K/V row to stage
  const int jq   = t & 3;    // quarter of the 64-dim head slice

  for (int jb = 0; jb < TK / 32; ++jb) {
    __syncthreads();   // all waves done reading previous K/V tile
    {
      const int jg = jb * 32 + jrow;
      int p, tok;
      if (jg < OTHERS) {                       // gathered tokens, p != img
        const int oi = jg / N_SEL;
        p   = oi + (oi >= img ? 1 : 0);
        tok = (int)indices[p * N_SEL + (jg % N_SEL)];
      } else {                                 // own tokens appended last
        p   = img;
        tok = jg - OTHERS;
      }
      const size_t gr = ((size_t)(b * IMG_COUNT + p)) * NUM_TOKENS + tok;
      // K row quarter: two async 16B copies straight into LDS
      const unsigned short* ksrc = Kb + gr * CHANNELS + (size_t)h * HD + jq * 16;
      const unsigned koff = (unsigned)((jrow * 72 + jq * 16) * 2);
      asyncLoadB128(ksBase + koff,      ksrc);
      asyncLoadB128(ksBase + koff + 16, ksrc + 8);
      // V row quarter: through VGPRs, transposed store
      union { uint4 q[2]; unsigned short s[16]; } vv;
      const unsigned short* vsrc = Vb + gr * CHANNELS + (size_t)h * HD + jq * 16;
      vv.q[0] = *(const uint4*)(vsrc);
      vv.q[1] = *(const uint4*)(vsrc + 8);
#pragma unroll
      for (int e = 0; e < 16; ++e) Vt[(jq * 16 + e) * 40 + jrow] = vv.s[e];
    }
    asm volatile("s_wait_asynccnt 0" ::: "memory");
    __syncthreads();

    // ---- S = Q (16x64) * K^T : two 16x16 tiles (j 0..15, 16..31) ----
    v8f s0, s1;
#pragma unroll
    for (int v = 0; v < 8; ++v) { s0[v] = 0.0f; s1[v] = 0.0f; }
    {
      BFrag kf;
      int off = lrow * 72 + (hiL ? 16 : 0);
      kf.q[0] = *(const uint4*)&Ks[off];      kf.q[1] = *(const uint4*)&Ks[off + 8];
      s0 = wmma_bf16(qa[0].v, kf.v, s0);
      kf.q[0] = *(const uint4*)&Ks[off + 32]; kf.q[1] = *(const uint4*)&Ks[off + 40];
      s0 = wmma_bf16(qa[1].v, kf.v, s0);
      off = (16 + lrow) * 72 + (hiL ? 16 : 0);
      kf.q[0] = *(const uint4*)&Ks[off];      kf.q[1] = *(const uint4*)&Ks[off + 8];
      s1 = wmma_bf16(qa[0].v, kf.v, s1);
      kf.q[0] = *(const uint4*)&Ks[off + 32]; kf.q[1] = *(const uint4*)&Ks[off + 40];
      s1 = wmma_bf16(qa[1].v, kf.v, s1);
    }

    // ---- online softmax (row reductions across 16-lane halves) ----
    float p0a[8], p1a[8], corr[8];
#pragma unroll
    for (int v = 0; v < 8; ++v) {
      float rm = fmaxf(s0[v], s1[v]);
      rm = fmaxf(rm, __shfl_xor(rm, 1, 32));
      rm = fmaxf(rm, __shfl_xor(rm, 2, 32));
      rm = fmaxf(rm, __shfl_xor(rm, 4, 32));
      rm = fmaxf(rm, __shfl_xor(rm, 8, 32));
      const float mn = fmaxf(m[v], rm);
      corr[v] = __expf(m[v] - mn);
      const float e0 = __expf(s0[v] - mn);
      const float e1 = __expf(s1[v] - mn);
      p0a[v] = e0; p1a[v] = e1;
      float rs = e0 + e1;
      rs += __shfl_xor(rs, 1, 32);
      rs += __shfl_xor(rs, 2, 32);
      rs += __shfl_xor(rs, 4, 32);
      rs += __shfl_xor(rs, 8, 32);
      l[v] = l[v] * corr[v] + rs;
      m[v] = mn;
    }
#pragma unroll
    for (int dt = 0; dt < 4; ++dt)
#pragma unroll
      for (int v = 0; v < 8; ++v) acc[dt][v] *= corr[v];

    // ---- repack P (C layout -> A layout) through per-wave LDS ----
    unsigned short* Pme = &Pw[wid * 16 * 40];
    const int prow = hiL ? 8 : 0;
#pragma unroll
    for (int v = 0; v < 8; ++v) {
      Pme[(prow + v) * 40 + lrow]      = f2bf(p0a[v]);
      Pme[(prow + v) * 40 + 16 + lrow] = f2bf(p1a[v]);
    }
    asm volatile("s_wait_dscnt 0" ::: "memory");   // in-wave LDS RAW
    BFrag pa;
    pa.q[0] = *(const uint4*)&Pme[lrow * 40 + ka];
    pa.q[1] = *(const uint4*)&Pme[lrow * 40 + ka + 8];

    // ---- O += P (16x32) * V (32x64) ----
#pragma unroll
    for (int dt = 0; dt < 4; ++dt) {
      BFrag vf;
      const int voff = (dt * 16 + lrow) * 40 + (hiL ? 16 : 0);
      vf.q[0] = *(const uint4*)&Vt[voff];
      vf.q[1] = *(const uint4*)&Vt[voff + 8];
      acc[dt] = wmma_bf16(pa.v, vf.v, acc[dt]);
    }
  }

  // ---- normalize and write O as bf16 (feeds final WMMA projection) ----
  const size_t orow0 = (size_t)bi * NUM_TOKENS + qblk * 64 + wid * 16 + (hiL ? 8 : 0);
#pragma unroll
  for (int v = 0; v < 8; ++v) {
    const float inv = 1.0f / l[v];
#pragma unroll
    for (int dt = 0; dt < 4; ++dt)
      Ob[(orow0 + v) * CHANNELS + (size_t)h * HD + dt * 16 + lrow] =
          f2bf(acc[dt][v] * inv);
  }
}

// ---------------------------------------------------------------------------
extern "C" void kernel_launch(void* const* d_in, const int* in_sizes, int n_in,
                              void* d_out, int out_size, void* d_ws, size_t ws_size,
                              hipStream_t stream) {
  const float*     hs  = (const float*)d_in[0];
  const float*     Wq  = (const float*)d_in[1];
  const float*     Wk  = (const float*)d_in[2];
  const float*     Wv  = (const float*)d_in[3];
  const float*     Wo  = (const float*)d_in[4];
  const float*     bo  = (const float*)d_in[5];
  const long long* idx = (const long long*)d_in[6];
  (void)in_sizes; (void)n_in; (void)out_size; (void)ws_size;

  const size_t nHS = (size_t)MTOT * CHANNELS;       // 7,372,800
  const size_t nW  = (size_t)CHANNELS * CHANNELS;   // 1,638,400

  unsigned short* hsb = (unsigned short*)d_ws;
  unsigned short* Wqb = hsb + nHS;
  unsigned short* Wkb = Wqb + nW;
  unsigned short* Wvb = Wkb + nW;
  unsigned short* Wob = Wvb + nW;
  unsigned short* Qb  = Wob + nW;
  unsigned short* Kb  = Qb + nHS;
  unsigned short* Vb  = Kb + nHS;
  unsigned short* Ob  = Vb + nHS;

  // one-time f32 -> bf16 conversions
  f32_to_bf16_kernel<<<4096, 256, 0, stream>>>(hs, hsb, (int)nHS);
  f32_to_bf16_kernel<<<1024, 256, 0, stream>>>(Wq, Wqb, (int)nW);
  f32_to_bf16_kernel<<<1024, 256, 0, stream>>>(Wk, Wkb, (int)nW);
  f32_to_bf16_kernel<<<1024, 256, 0, stream>>>(Wv, Wvb, (int)nW);
  f32_to_bf16_kernel<<<1024, 256, 0, stream>>>(Wo, Wob, (int)nW);

  const dim3 gg(MTOT / BM, CHANNELS / BN);
  // Q pre-scaled by 1/sqrt(64) = 0.125 (exact in bf16)
  gemm_bf16_async<<<gg, 256, 0, stream>>>(hsb, Wqb, Qb, nullptr, nullptr, 0.125f,
                                          CHANNELS, CHANNELS);
  gemm_bf16_async<<<gg, 256, 0, stream>>>(hsb, Wkb, Kb, nullptr, nullptr, 1.0f,
                                          CHANNELS, CHANNELS);
  gemm_bf16_async<<<gg, 256, 0, stream>>>(hsb, Wvb, Vb, nullptr, nullptr, 1.0f,
                                          CHANNELS, CHANNELS);

  const dim3 ga(NUM_TOKENS / 64, HEADS, 2 * IMG_COUNT);
  attn_kernel<<<ga, 128, 0, stream>>>(Qb, Kb, Vb, Ob, idx);

  gemm_bf16_async<<<gg, 256, 0, stream>>>(Ob, Wob, nullptr, (float*)d_out, bo, 1.0f,
                                          CHANNELS, CHANNELS);
}
